// TFLlamaAttention_58377195487582
// MI455X (gfx1250) — compile-verified, matrix operand order checked by reading, not run on hardware
//
#include <hip/hip_runtime.h>
#include <hip/hip_bf16.h>

// ---------------- constants ----------------
#define BATCH 2
#define SEQ   2048
#define NH    16
#define HD    128
#define HID   2048   // NH*HD
#define MTOT  4096   // BATCH*SEQ

typedef __bf16 v16bf __attribute__((ext_vector_type(16)));
typedef float  v8f   __attribute__((ext_vector_type(8)));
typedef unsigned int u32x4 __attribute__((ext_vector_type(4)));
typedef int          i32x4 __attribute__((ext_vector_type(4)));
typedef int          i32x8 __attribute__((ext_vector_type(8)));

union BF16x16 { v16bf v; unsigned u[8]; };

__device__ __forceinline__ unsigned short f32_to_bf16(float f) {
    union { float f; unsigned u; } x; x.f = f;
    unsigned r = x.u + 0x7FFFu + ((x.u >> 16) & 1u);   // round-to-nearest-even
    return (unsigned short)(r >> 16);
}
__device__ __forceinline__ float bf16_to_f32(unsigned short h) {
    union { unsigned u; float f; } x; x.u = ((unsigned)h) << 16;
    return x.f;
}

// K index of the first element of the bf16 pair held in 32-bit element v,
// for the given half-wave (ISA 16-bit A-matrix 16x32 layout; B mirrors with N).
__device__ __forceinline__ int kpair(int v, int half) {
    return (v < 4 ? 2 * v : 16 + 2 * (v - 4)) + 8 * half;
}

// ---------------- Tensor Data Mover: 2-D bf16 tile -> LDS ----------------
// D# per CDNA5 ISA. data_size = 2 bytes (code 1).
// Loads tile_d1 rows x tile_d0 elements from a [tensor_d1 x tensor_d0] tensor
// (row stride = stride0 elements) starting at gaddr, into LDS at lds_addr.
__device__ __forceinline__ void tdm_load_2d_bf16(unsigned lds_addr, const void* gaddr,
                                                 int tile_d0, int tile_d1,
                                                 int tensor_d0, int tensor_d1,
                                                 int stride0) {
    unsigned long long ga = (unsigned long long)(uintptr_t)gaddr;
    u32x4 g0;
    g0[0] = 1u;                                     // count=1, user mode
    g0[1] = lds_addr;                               // lds_addr (bytes)
    g0[2] = (unsigned)(ga & 0xFFFFFFFFu);           // global_addr[31:0]
    g0[3] = (unsigned)((ga >> 32) & 0x01FFFFFFu)    // global_addr[56:32]
            | 0x80000000u;                          // type=2 ("image")
    i32x8 g1;
    g1[0] = 0x10000;                                // data_size=1 (2 bytes)
    g1[1] = (int)(((unsigned)tensor_d0 & 0xFFFFu) << 16);          // td0[15:0] @63:48
    g1[2] = (int)((((unsigned)tensor_d0 >> 16) & 0xFFFFu)
            | (((unsigned)tensor_d1 & 0xFFFFu) << 16));            // td0 hi | td1 lo
    g1[3] = (int)((((unsigned)tensor_d1 >> 16) & 0xFFFFu)
            | (((unsigned)tile_d0 & 0xFFFFu) << 16));              // td1 hi | tile_d0
    g1[4] = (int)((unsigned)tile_d1 & 0xFFFFu);                    // tile_d1, tile_d2=0
    g1[5] = stride0;                                               // dim0_stride[31:0]
    g1[6] = 0;
    g1[7] = 0;
    i32x4 gz = {0, 0, 0, 0};
#if defined(__clang_major__) && (__clang_major__ >= 23)
    i32x8 gz8 = {0, 0, 0, 0, 0, 0, 0, 0};
    __builtin_amdgcn_tensor_load_to_lds(g0, g1, gz, gz, gz8, 0);
#else
    __builtin_amdgcn_tensor_load_to_lds(g0, g1, gz, gz, 0);
#endif
}

// ---------------- conversion kernels ----------------
__global__ void conv_f32_bf16_kernel(const float* __restrict__ in,
                                     unsigned short* __restrict__ out, int n) {
    for (int i = blockIdx.x * blockDim.x + threadIdx.x; i < n;
         i += gridDim.x * blockDim.x)
        out[i] = f32_to_bf16(in[i]);
}

// in: [rows x cols] f32 row-major ->  out: [cols x rows] bf16 row-major
__global__ __launch_bounds__(1024)
void transpose_conv_kernel(const float* __restrict__ in,
                           unsigned short* __restrict__ out,
                           int rows, int cols) {
    __shared__ float tile[32][33];
    int c = blockIdx.x * 32 + threadIdx.x;
    int r = blockIdx.y * 32 + threadIdx.y;
    tile[threadIdx.y][threadIdx.x] = in[(size_t)r * cols + c];
    __syncthreads();
    int tr = blockIdx.y * 32 + threadIdx.x;   // original row (k)
    int tc = blockIdx.x * 32 + threadIdx.y;   // original col (n)
    out[(size_t)tc * rows + tr] = f32_to_bf16(tile[threadIdx.x][threadIdx.y]);
}

// ---------------- WMMA GEMM ----------------
// C[M x N] = A[M x K] (bf16, row major) * Bt[N x K] (bf16, row major = B^T)
// MODE 0: write bf16 to [b,H,s,D]       (Q, K)
// MODE 1: write bf16 to [b,H,D,s]       (V transposed)
// MODE 2: write f32  to flat [M x N]    (final output)
template <int MODE>
__global__ __launch_bounds__(256)
void gemm_wmma_bf16(const unsigned short* __restrict__ A,
                    const unsigned short* __restrict__ Bt,
                    void* __restrict__ Cout, int M, int N, int K) {
    const int lane = threadIdx.x & 31;
    const int wid  = threadIdx.x >> 5;
    const int wm   = wid & 3;        // 4 waves along M
    const int wn   = wid >> 2;       // 2 waves along N
    const int m0   = blockIdx.y * 64 + wm * 16;
    const int n0   = blockIdx.x * 128 + wn * 64;
    const int mr   = lane & 15;
    const int half = lane >> 4;

    v8f acc[4] = {v8f{}, v8f{}, v8f{}, v8f{}};

    const unsigned short* Arow = A + (size_t)(m0 + mr) * K;
    for (int k0 = 0; k0 < K; k0 += 32) {
        BF16x16 a;
#pragma unroll
        for (int v = 0; v < 8; ++v)
            a.u[v] = *(const unsigned*)(Arow + k0 + kpair(v, half));
#pragma unroll
        for (int i = 0; i < 4; ++i) {
            const unsigned short* Brow = Bt + (size_t)(n0 + 16 * i + mr) * K;
            BF16x16 b;
#pragma unroll
            for (int v = 0; v < 8; ++v)
                b.u[v] = *(const unsigned*)(Brow + k0 + kpair(v, half));
            acc[i] = __builtin_amdgcn_wmma_f32_16x16x32_bf16(
                false, a.v, false, b.v, (short)0, acc[i], false, false);
        }
    }

#pragma unroll
    for (int i = 0; i < 4; ++i) {
#pragma unroll
        for (int r = 0; r < 8; ++r) {
            const int mrow = m0 + r + 8 * half;
            const int col  = n0 + 16 * i + mr;
            const float val = acc[i][r];
            if (MODE == 2) {
                ((float*)Cout)[(size_t)mrow * N + col] = val;
            } else {
                const int bb = mrow >> 11;       // / SEQ
                const int si = mrow & (SEQ - 1);
                const int h  = col >> 7;         // / HD
                const int d  = col & (HD - 1);
                size_t idx;
                if (MODE == 0)
                    idx = (((size_t)bb * NH + h) * SEQ + si) * HD + d;
                else
                    idx = (((size_t)bb * NH + h) * HD + d) * SEQ + si;
                ((unsigned short*)Cout)[idx] = f32_to_bf16(val);
            }
        }
    }
}

// ---------------- RoPE (in place, bf16) ----------------
__global__ void rope_kernel(unsigned short* __restrict__ q,
                            unsigned short* __restrict__ k,
                            const float* __restrict__ cosb,
                            const float* __restrict__ sinb, int total) {
    int idx = blockIdx.x * blockDim.x + threadIdx.x;
    if (idx >= total) return;
    const int d    = idx & 63;
    const int s    = (idx >> 6) & (SEQ - 1);
    const int rest = idx >> 17;                    // b*NH + h
    const size_t base = ((size_t)rest * SEQ + s) * HD;
    const float c1 = cosb[s * HD + d],      s1 = sinb[s * HD + d];
    const float c2 = cosb[s * HD + d + 64], s2 = sinb[s * HD + d + 64];

    float x1 = bf16_to_f32(q[base + d]);
    float x2 = bf16_to_f32(q[base + d + 64]);
    q[base + d]      = f32_to_bf16(x1 * c1 - x2 * s1);
    q[base + d + 64] = f32_to_bf16(x2 * c2 + x1 * s2);

    x1 = bf16_to_f32(k[base + d]);
    x2 = bf16_to_f32(k[base + d + 64]);
    k[base + d]      = f32_to_bf16(x1 * c1 - x2 * s1);
    k[base + d + 64] = f32_to_bf16(x2 * c2 + x1 * s2);
}

// ---------------- flash attention with TDM-staged K/V ----------------
// Q,K: [b,H,S,D] bf16 (post-RoPE);  Vt: [b,H,D,S] bf16
// Out: [b*S, HID] bf16 (attention output, pre-Wo)
// grid (SEQ/128, NH, BATCH), block 256 (8 waves, one 16-query tile per wave).
// K/V tiles for the whole block are DMA'd into double-buffered LDS by the
// Tensor Data Mover (one tensor_load_to_lds per tile, issued by wave 0),
// overlapped with compute on the other buffer.
__global__ __launch_bounds__(256)
void flash_attn_kernel(const unsigned short* __restrict__ Qb,
                       const unsigned short* __restrict__ Kb,
                       const unsigned short* __restrict__ Vt,
                       unsigned short* __restrict__ Out) {
    __shared__ unsigned short ldsK[2][32 * 128];   // [key][d]   8KB x2
    __shared__ unsigned short ldsV[2][128 * 32];   // [d][key]   8KB x2
    __shared__ unsigned short ldsP[8 * 16 * 32];   // per-wave P tile 8KB

    const int lane = threadIdx.x & 31;
    const int wid  = threadIdx.x >> 5;
    const int h    = blockIdx.y;
    const int bb   = blockIdx.z;
    const int q0   = blockIdx.x * 128 + wid * 16;
    const int mr   = lane & 15;
    const int half = lane >> 4;
    const float scale = 0.088388347648318447f;  // 1/sqrt(128)

    const size_t headQK = ((size_t)bb * NH + h) * (size_t)SEQ * HD;
    const size_t headV  = ((size_t)bb * NH + h) * (size_t)HD * SEQ;
    const unsigned short* Khead = Kb + headQK;
    const unsigned short* Vhead = Vt + headV;

    // Q tile resident in registers (A-operand fragments, 4 K-steps of 32)
    const unsigned short* qrow = Qb + headQK + (size_t)(q0 + mr) * HD;
    BF16x16 qa[4];
#pragma unroll
    for (int t = 0; t < 4; ++t)
#pragma unroll
        for (int v = 0; v < 8; ++v)
            qa[t].u[v] = *(const unsigned*)(qrow + t * 32 + kpair(v, half));

    v8f O[8] = {v8f{}, v8f{}, v8f{}, v8f{}, v8f{}, v8f{}, v8f{}, v8f{}};
    float mstat[8], lstat[8];
#pragma unroll
    for (int r = 0; r < 8; ++r) { mstat[r] = -1e30f; lstat[r] = 0.0f; }

    unsigned short* myP = ldsP + wid * (16 * 32);

    // prologue: DMA first K/V tile into buffer 0
    if (wid == 0) {
        tdm_load_2d_bf16((unsigned)(uintptr_t)&ldsK[0][0], Khead,
                         /*tile_d0=*/HD, /*tile_d1=*/32,
                         /*tensor_d0=*/HD, /*tensor_d1=*/SEQ, /*stride0=*/HD);
        tdm_load_2d_bf16((unsigned)(uintptr_t)&ldsV[0][0], Vhead,
                         /*tile_d0=*/32, /*tile_d1=*/HD,
                         /*tensor_d0=*/SEQ, /*tensor_d1=*/HD, /*stride0=*/SEQ);
    }

    int cur = 0;
    for (int j = 0; j < SEQ; j += 32) {
        // issue next tile's DMA into the free buffer, then wait for current
        if (wid == 0) {
            if (j + 32 < SEQ) {
                tdm_load_2d_bf16((unsigned)(uintptr_t)&ldsK[cur ^ 1][0],
                                 Khead + (size_t)(j + 32) * HD,
                                 HD, 32, HD, SEQ, HD);
                tdm_load_2d_bf16((unsigned)(uintptr_t)&ldsV[cur ^ 1][0],
                                 Vhead + (j + 32),
                                 32, HD, SEQ, HD, SEQ);
                __builtin_amdgcn_s_wait_tensorcnt(2);  // current buffer landed
            } else {
                __builtin_amdgcn_s_wait_tensorcnt(0);
            }
        }
        __syncthreads();

        const unsigned short* kbuf = &ldsK[cur][0];
        const unsigned short* vbuf = &ldsV[cur][0];

        // ---- scores: two 16x16 tiles over keys [j, j+32) ----
        v8f s0 = v8f{}, s1 = v8f{};
#pragma unroll
        for (int t = 0; t < 4; ++t) {
            BF16x16 kb;
#pragma unroll
            for (int v = 0; v < 8; ++v)
                kb.u[v] = *(const unsigned*)(kbuf + mr * HD + t * 32 + kpair(v, half));
            s0 = __builtin_amdgcn_wmma_f32_16x16x32_bf16(
                false, qa[t].v, false, kb.v, (short)0, s0, false, false);
        }
#pragma unroll
        for (int t = 0; t < 4; ++t) {
            BF16x16 kb;
#pragma unroll
            for (int v = 0; v < 8; ++v)
                kb.u[v] = *(const unsigned*)(kbuf + (16 + mr) * HD + t * 32 + kpair(v, half));
            s1 = __builtin_amdgcn_wmma_f32_16x16x32_bf16(
                false, qa[t].v, false, kb.v, (short)0, s1, false, false);
        }

        // ---- online softmax (per-row across 16 lanes of each half) ----
        float alpha[8];
#pragma unroll
        for (int r = 0; r < 8; ++r) {
            const float a = s0[r] * scale;
            const float b = s1[r] * scale;
            float rm = fmaxf(a, b);
            rm = fmaxf(rm, __shfl_xor(rm, 1));
            rm = fmaxf(rm, __shfl_xor(rm, 2));
            rm = fmaxf(rm, __shfl_xor(rm, 4));
            rm = fmaxf(rm, __shfl_xor(rm, 8));
            const float mnew = fmaxf(mstat[r], rm);
            alpha[r] = __expf(mstat[r] - mnew);
            mstat[r] = mnew;
            const float p0 = __expf(a - mnew);
            const float p1 = __expf(b - mnew);
            lstat[r] = lstat[r] * alpha[r] + p0 + p1;  // lane-partial row sum
            myP[(r + 8 * half) * 32 + mr]      = f32_to_bf16(p0);
            myP[(r + 8 * half) * 32 + 16 + mr] = f32_to_bf16(p1);
        }
#pragma unroll
        for (int nt = 0; nt < 8; ++nt)
#pragma unroll
            for (int r = 0; r < 8; ++r)
                O[nt][r] *= alpha[r];

        // wave-local C-layout -> A-layout relayout through LDS
        asm volatile("s_wait_dscnt 0" ::: "memory");
        BF16x16 pa;
#pragma unroll
        for (int v = 0; v < 8; ++v)
            pa.u[v] = *(const unsigned*)(myP + mr * 32 + kpair(v, half));

        // ---- O += P (16x32) * V (32x128) ----
#pragma unroll
        for (int nt = 0; nt < 8; ++nt) {
            BF16x16 vb;
#pragma unroll
            for (int v = 0; v < 8; ++v)
                vb.u[v] = *(const unsigned*)(vbuf + (nt * 16 + mr) * 32 + kpair(v, half));
            O[nt] = __builtin_amdgcn_wmma_f32_16x16x32_bf16(
                false, pa.v, false, vb.v, (short)0, O[nt], false, false);
        }

        asm volatile("s_wait_dscnt 0" ::: "memory");  // all my LDS reads done
        __syncthreads();                              // whole block done with buffers
        cur ^= 1;
    }

    // ---- finalize: reduce l across the 16-lane half, normalize, store ----
    float inv[8];
#pragma unroll
    for (int r = 0; r < 8; ++r) {
        float l = lstat[r];
        l += __shfl_xor(l, 1);
        l += __shfl_xor(l, 2);
        l += __shfl_xor(l, 4);
        l += __shfl_xor(l, 8);
        inv[r] = 1.0f / l;
    }
#pragma unroll
    for (int nt = 0; nt < 8; ++nt)
#pragma unroll
        for (int r = 0; r < 8; ++r) {
            const int row = bb * SEQ + q0 + r + 8 * half;
            const int col = h * HD + nt * 16 + mr;
            Out[(size_t)row * HID + col] = f32_to_bf16(O[nt][r] * inv[r]);
        }
}

// ---------------- launch ----------------
extern "C" void kernel_launch(void* const* d_in, const int* in_sizes, int n_in,
                              void* d_out, int out_size, void* d_ws, size_t ws_size,
                              hipStream_t stream) {
    (void)in_sizes; (void)n_in; (void)out_size; (void)ws_size;
    const float* hs  = (const float*)d_in[0];
    const float* cs  = (const float*)d_in[1];
    const float* sn  = (const float*)d_in[2];
    const float* Wq  = (const float*)d_in[3];
    const float* Wk  = (const float*)d_in[4];
    const float* Wv  = (const float*)d_in[5];
    const float* Wo  = (const float*)d_in[6];
    float* out = (float*)d_out;

    char* ws = (char*)d_ws;
    size_t o = 0;
    auto alloc = [&](size_t bytes) { char* p = ws + o; o += (bytes + 255) & ~(size_t)255; return p; };
    unsigned short* hsb = (unsigned short*)alloc((size_t)MTOT * HID * 2);
    unsigned short* wqt = (unsigned short*)alloc((size_t)HID * HID * 2);
    unsigned short* wkt = (unsigned short*)alloc((size_t)HID * HID * 2);
    unsigned short* wvt = (unsigned short*)alloc((size_t)HID * HID * 2);
    unsigned short* wot = (unsigned short*)alloc((size_t)HID * HID * 2);
    unsigned short* qb  = (unsigned short*)alloc((size_t)BATCH * NH * SEQ * HD * 2);
    unsigned short* kb  = (unsigned short*)alloc((size_t)BATCH * NH * SEQ * HD * 2);
    unsigned short* vtb = (unsigned short*)alloc((size_t)BATCH * NH * SEQ * HD * 2);
    unsigned short* aob = (unsigned short*)alloc((size_t)MTOT * HID * 2);

    // 1. conversions
    {
        int n = MTOT * HID;
        conv_f32_bf16_kernel<<<4096, 256, 0, stream>>>(hs, hsb, n);
        dim3 tb(32, 32), tg(HID / 32, HID / 32);
        transpose_conv_kernel<<<tg, tb, 0, stream>>>(Wq, wqt, HID, HID);
        transpose_conv_kernel<<<tg, tb, 0, stream>>>(Wk, wkt, HID, HID);
        transpose_conv_kernel<<<tg, tb, 0, stream>>>(Wv, wvt, HID, HID);
        transpose_conv_kernel<<<tg, tb, 0, stream>>>(Wo, wot, HID, HID);
    }
    // 2. q/k/v projections
    {
        dim3 g(HID / 128, MTOT / 64);
        gemm_wmma_bf16<0><<<g, 256, 0, stream>>>(hsb, wqt, qb,  MTOT, HID, HID);
        gemm_wmma_bf16<0><<<g, 256, 0, stream>>>(hsb, wkt, kb,  MTOT, HID, HID);
        gemm_wmma_bf16<1><<<g, 256, 0, stream>>>(hsb, wvt, vtb, MTOT, HID, HID);
    }
    // 3. RoPE
    {
        int total = BATCH * NH * SEQ * 64;
        rope_kernel<<<(total + 255) / 256, 256, 0, stream>>>(qb, kb, cs, sn, total);
    }
    // 4. flash attention
    {
        dim3 g(SEQ / 128, NH, BATCH);
        flash_attn_kernel<<<g, 256, 0, stream>>>(qb, kb, vtb, aob);
    }
    // 5. output projection -> fp32 d_out
    {
        dim3 g(HID / 128, MTOT / 64);
        gemm_wmma_bf16<2><<<g, 256, 0, stream>>>(aob, wot, out, MTOT, HID, HID);
    }
}
